// LightningProBlock_21268678049968
// MI455X (gfx1250) — matrix operand order repaired and checked by reading, not compiled, not run on hardware
//
#include <hip/hip_runtime.h>

// ---------------------------------------------------------------------------
// LightningProBlock for MI455X (gfx1250, wave32, WMMA).
// All dense projections go through a generic f16-WMMA GEMM (f32 accumulate).
// ---------------------------------------------------------------------------

typedef __attribute__((ext_vector_type(16))) _Float16 v16h;
typedef __attribute__((ext_vector_type(8)))  _Float16 v8h;
typedef __attribute__((ext_vector_type(8)))  float    v8f;

constexpr int B_   = 2;
constexpr int C_   = 256;
constexpr int H_   = 48;
constexpr int W_   = 48;
constexpr int N_   = H_ * W_;     // 2304
constexpr int NH_  = 8;
constexpr int HD_  = 32;          // head dim
constexpr int KK_  = 1152;        // top-k count = int(N*0.5)
constexpr int GH_  = 11;          // window grid
constexpr int GWID_= 11;
constexpr int NWIN_= GH_ * GWID_; // 121
constexpr int WT_  = 64;          // tokens per window
constexpr int RD_  = 93;
constexpr int NB_  = 31;
constexpr int FC_  = 3 * RD_;     // 279 fused channels

// ---------------------------------------------------------------------------
// Generic GEMM: out[M,N] = A[M,K] @ W[N,K]^T (+bias[N]).  fp32 in/out.
// Block tile 64x64, 256 threads = 8 waves; each wave owns a 16x32 strip
// (two 16x16 WMMA accumulators -> two v_wmma issues per K-step).
// Interior tiles (the dominant case for M=4608/7744, N=768/256, K=256) use a
// branchless float4 (global_load_b128) + ds_store_b128 staging path; edge
// tiles / unaligned-K fall back to guarded scalar loads with zero padding.
// Fragments use contiguous k = half*16 + i in both A and B (16-bit A 16x32 / B 32x16).
// ---------------------------------------------------------------------------
__global__ __launch_bounds__(256)
void gemm_wmma_kernel(const float* __restrict__ A, const float* __restrict__ Wt,
                      const float* __restrict__ bias, float* __restrict__ out,
                      int M, int N, int K) {
  __shared__ _Float16 sA[64][32];   // [m_local][k_local]
  __shared__ _Float16 sB[64][32];   // [n_local][k_local]
  const int tid  = threadIdx.x;
  const int lane = tid & 31;
  const int wave = tid >> 5;                 // 0..7
  const int ti   = (wave >> 1) * 16;         // M sub-tile offset (0/16/32/48)
  const int nj   = (wave & 1) * 32;          // N strip offset (0/32)
  const int m0   = blockIdx.y * 64;
  const int n0   = blockIdx.x * 64;
  const int half = lane >> 4;                // 0 | 1
  const int l16  = lane & 15;

  // staging assignment: thread -> row r (0..63), 8 contiguous k at c0
  const int r  = tid >> 2;
  const int c0 = (tid & 3) * 8;

  const bool interior = (m0 + 64 <= M) && (n0 + 64 <= N);
  const bool kalign   = (K & 3) == 0;

  v8f acc0 = {};
  v8f acc1 = {};

  for (int k0 = 0; k0 < K; k0 += 32) {
    if (interior && kalign && (k0 + 32 <= K)) {
      // ---- fast path: vector loads, no predication ----
      const float* pa = A  + (size_t)(m0 + r) * K + k0 + c0;
      const float* pb = Wt + (size_t)(n0 + r) * K + k0 + c0;
      float4 a0 = *(const float4*)pa;
      float4 a1 = *(const float4*)(pa + 4);
      float4 b0 = *(const float4*)pb;
      float4 b1 = *(const float4*)(pb + 4);
      v8h ha, hb;
      ha[0] = (_Float16)a0.x; ha[1] = (_Float16)a0.y; ha[2] = (_Float16)a0.z; ha[3] = (_Float16)a0.w;
      ha[4] = (_Float16)a1.x; ha[5] = (_Float16)a1.y; ha[6] = (_Float16)a1.z; ha[7] = (_Float16)a1.w;
      hb[0] = (_Float16)b0.x; hb[1] = (_Float16)b0.y; hb[2] = (_Float16)b0.z; hb[3] = (_Float16)b0.w;
      hb[4] = (_Float16)b1.x; hb[5] = (_Float16)b1.y; hb[6] = (_Float16)b1.z; hb[7] = (_Float16)b1.w;
      *(v8h*)&sA[r][c0] = ha;
      *(v8h*)&sB[r][c0] = hb;
    } else {
      // ---- edge path: guarded scalar loads, zero pad ----
      const int gm = m0 + r;
      const int gn = n0 + r;
#pragma unroll
      for (int e = 0; e < 8; ++e) {
        int gk = k0 + c0 + e;
        float va = (gm < M && gk < K) ? A[(size_t)gm * K + gk] : 0.0f;
        sA[r][c0 + e] = (_Float16)va;
        float vb = (gn < N && gk < K) ? Wt[(size_t)gn * K + gk] : 0.0f;
        sB[r][c0 + e] = (_Float16)vb;
      }
    }
    __syncthreads();

    v16h af, bf0, bf1;
#pragma unroll
    for (int i = 0; i < 16; ++i) {
      // A 16x32 f16 layout: K = half*16 + i, row = lane%16
      int ka = half * 16 + i;
      af[i] = sA[ti + l16][ka];
      // B 32x16 f16 layout: K = half*16 + i, column = lane%16
      int kb = half * 16 + i;
      bf0[i] = sB[nj + l16][kb];
      bf1[i] = sB[nj + 16 + l16][kb];
    }
    acc0 = __builtin_amdgcn_wmma_f32_16x16x32_f16(false, af, false, bf0,
                                                  (short)0, acc0, false, false);
    acc1 = __builtin_amdgcn_wmma_f32_16x16x32_f16(false, af, false, bf1,
                                                  (short)0, acc1, false, false);
    __syncthreads();
  }

  // C/D layout: VGPR r, lanes 0-15 -> M=r, lanes 16-31 -> M=8+r; N = lane%16
#pragma unroll
  for (int rr = 0; rr < 8; ++rr) {
    int gm = m0 + ti + half * 8 + rr;
    int gn0 = n0 + nj + l16;
    int gn1 = gn0 + 16;
    if (gm < M && gn0 < N) {
      float v = acc0[rr];
      if (bias) v += bias[gn0];
      out[(size_t)gm * N + gn0] = v;
    }
    if (gm < M && gn1 < N) {
      float v = acc1[rr];
      if (bias) v += bias[gn1];
      out[(size_t)gm * N + gn1] = v;
    }
  }
}

// ---------------------------------------------------------------------------
// (B,C,H,W) -> tokens (B*N, C)
// ---------------------------------------------------------------------------
__global__ __launch_bounds__(256)
void transpose_nc_kernel(const float* __restrict__ x, float* __restrict__ t) {
  int idx = blockIdx.x * 256 + threadIdx.x;           // ((b*N+n)*C + c)
  if (idx >= B_ * N_ * C_) return;
  int c = idx & (C_ - 1);
  int n = (idx / C_) % N_;
  int b = idx / (C_ * N_);
  t[idx] = x[((size_t)(b * C_ + c)) * N_ + n];
}

// ---------------------------------------------------------------------------
// Sparse attention core: one block = one (b, head, query).  Scores row in
// LDS, exact kk-th-largest via bitwise radix descend on order-preserving
// uint keys (== jax top_k threshold), masked softmax, attn @ V.
// ---------------------------------------------------------------------------
__device__ inline unsigned ordf(float f) {
  unsigned u = __float_as_uint(f);
  return (u & 0x80000000u) ? ~u : (u | 0x80000000u);
}

__global__ __launch_bounds__(256)
void sparse_attn_kernel(const float* __restrict__ qkv,
                        const float* __restrict__ temperature,
                        float* __restrict__ attn_out) {
  __shared__ float sc[N_];
  __shared__ float qv[HD_];
  __shared__ int   redi[256];
  __shared__ float redf[256];

  const int blk = blockIdx.x;
  const int n = blk % N_;
  const int h = (blk / N_) % NH_;
  const int b = blk / (N_ * NH_);
  const int tid = threadIdx.x;

  const float temp = fmaxf(temperature[0], 0.01f);
  const float scale = 0.17677669529663687f / temp;   // hd^-0.5 / temp
  const size_t rowQ = ((size_t)(b * N_ + n)) * (3 * C_);

  if (tid < HD_) qv[tid] = qkv[rowQ + h * HD_ + tid];
  __syncthreads();

  // scores
  for (int m = tid; m < N_; m += 256) {
    const float* kr = qkv + ((size_t)(b * N_ + m)) * (3 * C_) + C_ + h * HD_;
    float s = 0.0f;
#pragma unroll
    for (int d = 0; d < HD_; ++d) s += qv[d] * kr[d];
    sc[m] = s * scale;
  }
  __syncthreads();

  // exact kk-th largest (threshold) via 32-step radix descend
  unsigned prefix = 0u;
  for (int bit = 31; bit >= 0; --bit) {
    unsigned cand = prefix | (1u << bit);
    int cnt = 0;
    for (int m = tid; m < N_; m += 256) cnt += (ordf(sc[m]) >= cand) ? 1 : 0;
    redi[tid] = cnt;
    __syncthreads();
    for (int s = 128; s > 0; s >>= 1) {
      if (tid < s) redi[tid] += redi[tid + s];
      __syncthreads();
    }
    int total = redi[0];
    __syncthreads();
    if (total >= KK_) prefix = cand;
  }

  // masked row max
  float mx = -3.4e38f;
  for (int m = tid; m < N_; m += 256)
    if (ordf(sc[m]) >= prefix) mx = fmaxf(mx, sc[m]);
  redf[tid] = mx;
  __syncthreads();
  for (int s = 128; s > 0; s >>= 1) {
    if (tid < s) redf[tid] = fmaxf(redf[tid], redf[tid + s]);
    __syncthreads();
  }
  mx = redf[0];
  __syncthreads();

  // exp + sum (masked entries -> 0 probability)
  float lsum = 0.0f;
  for (int m = tid; m < N_; m += 256) {
    float e = (ordf(sc[m]) >= prefix) ? __expf(sc[m] - mx) : 0.0f;
    sc[m] = e;
    lsum += e;
  }
  redf[tid] = lsum;
  __syncthreads();
  for (int s = 128; s > 0; s >>= 1) {
    if (tid < s) redf[tid] += redf[tid + s];
    __syncthreads();
  }
  const float inv = 1.0f / redf[0];
  __syncthreads();

  // out[d] = sum_m p[m] * V[m,d]   (8 chunks x 32 dims)
  const int d = tid & 31, chunk = tid >> 5;
  float part = 0.0f;
  for (int m = chunk; m < N_; m += 8)
    part += sc[m] * qkv[((size_t)(b * N_ + m)) * (3 * C_) + 2 * C_ + h * HD_ + d];
  redf[tid] = part;
  __syncthreads();
  if (tid < HD_) {
    float s = 0.0f;
#pragma unroll
    for (int c2 = 0; c2 < 8; ++c2) s += redf[c2 * 32 + tid];
    attn_out[((size_t)(b * N_ + n)) * C_ + h * HD_ + tid] = s * inv;
  }
}

// x1 = x + proj^T  (tokens back to BCHW)
__global__ __launch_bounds__(256)
void add_residual_t_kernel(const float* __restrict__ x,
                           const float* __restrict__ proj,
                           float* __restrict__ x1) {
  int idx = blockIdx.x * 256 + threadIdx.x;           // ((b*C+c)*N + n)
  if (idx >= B_ * C_ * N_) return;
  int n = idx % N_;
  int c = (idx / N_) % C_;
  int b = idx / (N_ * C_);
  x1[idx] = x[idx] + proj[((size_t)(b * N_ + n)) * C_ + c];
}

// ---------------------------------------------------------------------------
// Window attention (one batch element per launch group)
// ---------------------------------------------------------------------------
__global__ __launch_bounds__(256)
void gather_windows_kernel(const float* __restrict__ x1, float* __restrict__ t,
                           int b) {
  int idx = blockIdx.x * 256 + threadIdx.x;           // ((g*64+i)*C + c)
  if (idx >= NWIN_ * WT_ * C_) return;
  int c = idx & (C_ - 1);
  int i = (idx / C_) & (WT_ - 1);
  int g = idx / (C_ * WT_);
  int gh = g / GWID_, gw = g % GWID_;
  int wi = i >> 3, wj = i & 7;
  int ih = gh * 4 + wi, iw = gw * 4 + wj;
  t[idx] = x1[(((size_t)(b * C_ + c)) * H_ + ih) * W_ + iw];
}

__global__ __launch_bounds__(256)
void window_attn_kernel(const float* __restrict__ qkv, float* __restrict__ attn_out) {
  __shared__ float qs[WT_][HD_], ks[WT_][HD_], vs[WT_][HD_], pr[WT_][WT_];
  const int g = blockIdx.x >> 3, h = blockIdx.x & 7;
  const int tid = threadIdx.x;

  for (int e = tid; e < WT_ * HD_; e += 256) {
    int i = e >> 5, d = e & 31;
    const float* row = qkv + ((size_t)(g * WT_ + i)) * (3 * C_);
    qs[i][d] = row[h * HD_ + d];
    ks[i][d] = row[C_ + h * HD_ + d];
    vs[i][d] = row[2 * C_ + h * HD_ + d];
  }
  __syncthreads();

  const float scale = 0.17677669529663687f;           // hd^-0.5
  for (int e = tid; e < WT_ * WT_; e += 256) {
    int i = e >> 6, j = e & 63;
    float s = 0.0f;
#pragma unroll
    for (int d = 0; d < HD_; ++d) s += qs[i][d] * ks[j][d];
    pr[i][j] = s * scale;
  }
  __syncthreads();

  if (tid < WT_) {
    float mx = -3.4e38f;
    for (int j = 0; j < WT_; ++j) mx = fmaxf(mx, pr[tid][j]);
    float sum = 0.0f;
    for (int j = 0; j < WT_; ++j) { float e2 = __expf(pr[tid][j] - mx); pr[tid][j] = e2; sum += e2; }
    float inv = 1.0f / sum;
    for (int j = 0; j < WT_; ++j) pr[tid][j] *= inv;
  }
  __syncthreads();

  for (int e = tid; e < WT_ * HD_; e += 256) {
    int i = e >> 5, d = e & 31;
    float o = 0.0f;
    for (int j = 0; j < WT_; ++j) o += pr[i][j] * vs[j][d];
    attn_out[((size_t)(g * WT_ + i)) * C_ + h * HD_ + d] = o;
  }
}

// overlap-average scatter + residual:  x2 = x1 + sum(win)/cnt
__global__ __launch_bounds__(256)
void window_scatter_kernel(const float* __restrict__ x1,
                           const float* __restrict__ proj,
                           float* __restrict__ x2, int b) {
  int idx = blockIdx.x * 256 + threadIdx.x;           // (c*H + y)*W + x
  if (idx >= C_ * H_ * W_) return;
  int xw = idx % W_;
  int y  = (idx / W_) % H_;
  int c  = idx / (W_ * H_);
  int ahlo = (y >= 7) ? ((y - 4) >> 2) : 0;           // ceil((y-7)/4)
  int ahhi = min(y >> 2, GH_ - 1);
  int awlo = (xw >= 7) ? ((xw - 4) >> 2) : 0;
  int awhi = min(xw >> 2, GWID_ - 1);
  float s = 0.0f; int cnt = 0;
  for (int ah = ahlo; ah <= ahhi; ++ah)
    for (int aw = awlo; aw <= awhi; ++aw) {
      int wi = y - ah * 4, wj = xw - aw * 4;
      int g = ah * GWID_ + aw, i = wi * 8 + wj;
      s += proj[((size_t)(g * WT_ + i)) * C_ + c];
      ++cnt;
    }
  size_t o = (((size_t)(b * C_ + c)) * H_ + y) * W_ + xw;
  x2[o] = x1[o] + s / (float)max(cnt, 1);
}

// ---------------------------------------------------------------------------
// Spectral branch
// ---------------------------------------------------------------------------
__global__ __launch_bounds__(256)
void pool_kernel(const float* __restrict__ x2, float* __restrict__ xp,
                 int s) {
  int ss = s * s, hs = H_ / s;
  int idx = blockIdx.x * 256 + threadIdx.x;           // ((b*ss+t)*C + c)
  if (idx >= B_ * ss * C_) return;
  int c = idx & (C_ - 1);
  int t = (idx / C_) % ss;
  int b = idx / (C_ * ss);
  int i = t / s, j = t % s;
  float sum = 0.0f;
  for (int p = 0; p < hs; ++p)
    for (int q = 0; q < hs; ++q)
      sum += x2[(((size_t)(b * C_ + c)) * H_ + i * hs + p) * W_ + j * hs + q];
  xp[idx] = sum / (float)(hs * hs);
}

__global__ __launch_bounds__(128)
void spec_attn_kernel(const float* __restrict__ qkvp,
                      const float* __restrict__ sw,
                      float* __restrict__ ospec, int ss) {
  __shared__ float q_[RD_], k_[RD_], v_[RD_], qs2[RD_], logit[9], att[9];
  const int blk = blockIdx.x;                          // b*ss + t
  const int t = blk % ss, b = blk / ss;
  const int tid = threadIdx.x;
  const float* row = qkvp + ((size_t)(b * ss + t)) * FC_;

  if (tid < RD_) { q_[tid] = row[tid]; k_[tid] = row[RD_ + tid]; v_[tid] = row[2 * RD_ + tid]; }
  __syncthreads();

  if (tid < RD_) {                                     // qs = q' @ sw
    int g = tid / NB_, c = tid % NB_;
    float s = 0.0f;
    for (int nb = 0; nb < NB_; ++nb) s += q_[g * NB_ + nb] * sw[nb * NB_ + c];
    qs2[tid] = s;
  }
  __syncthreads();

  if (tid < 9) {                                       // logits = qs @ k' * 3^-0.5
    int g = tid / 3, g2 = tid % 3;
    float s = 0.0f;
    for (int nb = 0; nb < NB_; ++nb) s += qs2[g * NB_ + nb] * k_[g2 * NB_ + nb];
    logit[tid] = s * 0.5773502691896258f;
  }
  __syncthreads();

  if (tid < 3) {                                       // softmax over g2 (size 3)
    float m0 = fmaxf(logit[tid * 3], fmaxf(logit[tid * 3 + 1], logit[tid * 3 + 2]));
    float e0 = __expf(logit[tid * 3] - m0);
    float e1 = __expf(logit[tid * 3 + 1] - m0);
    float e2 = __expf(logit[tid * 3 + 2] - m0);
    float inv = 1.0f / (e0 + e1 + e2);
    att[tid * 3] = e0 * inv; att[tid * 3 + 1] = e1 * inv; att[tid * 3 + 2] = e2 * inv;
  }
  __syncthreads();

  if (tid < RD_) {                                     // o = attn @ v'
    int g = tid / NB_, nb = tid % NB_;
    float o = att[g * 3 + 0] * v_[0 * NB_ + nb]
            + att[g * 3 + 1] * v_[1 * NB_ + nb]
            + att[g * 3 + 2] * v_[2 * NB_ + nb];
    ospec[((size_t)(b * RD_ + tid)) * ss + t] = o;
  }
}

// bilinear (half-pixel, clamped) upsample (s,s)->(48,48), write channel slice
__global__ __launch_bounds__(256)
void resize_kernel(const float* __restrict__ ospec, float* __restrict__ fused,
                   int s, int poff) {
  int ss = s * s;
  int idx = blockIdx.x * 256 + threadIdx.x;           // ((b*RD+r)*H + y)*W + x
  if (idx >= B_ * RD_ * H_ * W_) return;
  int xw = idx % W_;
  int y  = (idx / W_) % H_;
  int r  = (idx / (W_ * H_)) % RD_;
  int b  = idx / (W_ * H_ * RD_);
  float ratio = (float)s / (float)H_;
  float sy = fmaxf((y + 0.5f) * ratio - 0.5f, 0.0f);
  float sx = fmaxf((xw + 0.5f) * ratio - 0.5f, 0.0f);
  int y0 = min((int)sy, s - 1), x0 = min((int)sx, s - 1);
  int y1 = min(y0 + 1, s - 1), x1 = min(x0 + 1, s - 1);
  float fy = sy - (float)y0, fx = sx - (float)x0;
  const float* src = ospec + ((size_t)(b * RD_ + r)) * ss;
  float v00 = src[y0 * s + x0], v01 = src[y0 * s + x1];
  float v10 = src[y1 * s + x0], v11 = src[y1 * s + x1];
  float v = (1.0f - fy) * ((1.0f - fx) * v00 + fx * v01)
          + fy * ((1.0f - fx) * v10 + fx * v11);
  fused[((size_t)(b * N_ + y * W_ + xw)) * FC_ + poff + r] = v;
}

// y = gemm_out^T + x2
__global__ __launch_bounds__(256)
void fuse_residual_kernel(const float* __restrict__ gout,
                          const float* __restrict__ x2,
                          float* __restrict__ ybuf) {
  int idx = blockIdx.x * 256 + threadIdx.x;           // ((b*C+c)*N + n)
  if (idx >= B_ * C_ * N_) return;
  int n = idx % N_;
  int c = (idx / N_) % C_;
  int b = idx / (N_ * C_);
  ybuf[idx] = gout[((size_t)(b * N_ + n)) * C_ + c] + x2[idx];
}

// GroupNorm stats: one block per (b, group); 8 channels x 2304 pixels
__global__ __launch_bounds__(256)
void gn_stats_kernel(const float* __restrict__ ybuf, float* __restrict__ stats) {
  __shared__ float rs[256], rq[256];
  const int blk = blockIdx.x;                          // b*32 + g
  const int b = blk / 32, g = blk % 32;
  const int tid = threadIdx.x;
  const int total = 8 * N_;
  float s = 0.0f, q = 0.0f;
  for (int e = tid; e < total; e += 256) {
    int c2 = e / N_, n = e % N_;
    float v = ybuf[((size_t)(b * C_ + g * 8 + c2)) * N_ + n];
    s += v; q += v * v;
  }
  rs[tid] = s; rq[tid] = q;
  __syncthreads();
  for (int st = 128; st > 0; st >>= 1) {
    if (tid < st) { rs[tid] += rs[tid + st]; rq[tid] += rq[tid + st]; }
    __syncthreads();
  }
  if (tid == 0) {
    float mu = rs[0] / (float)total;
    float var = rq[0] / (float)total - mu * mu;
    stats[blk * 2] = mu;
    stats[blk * 2 + 1] = var;
  }
}

__global__ __launch_bounds__(256)
void gn_apply_kernel(const float* __restrict__ ybuf, const float* __restrict__ stats,
                     const float* __restrict__ gamma, const float* __restrict__ beta,
                     float* __restrict__ out) {
  int idx = blockIdx.x * 256 + threadIdx.x;           // ((b*C+c)*N + n)
  if (idx >= B_ * C_ * N_) return;
  int c = (idx / N_) % C_;
  int b = idx / (N_ * C_);
  int g = c >> 3;
  float mu = stats[(b * 32 + g) * 2];
  float var = stats[(b * 32 + g) * 2 + 1];
  out[idx] = (ybuf[idx] - mu) * rsqrtf(var + 1e-5f) * gamma[c] + beta[c];
}

// ---------------------------------------------------------------------------
// Host side
// ---------------------------------------------------------------------------
static inline int cdiv(int a, int b) { return (a + b - 1) / b; }

static void launch_gemm(const float* A, const float* Wt, const float* bias,
                        float* out, int M, int N, int K, hipStream_t stream) {
  dim3 grid(cdiv(N, 64), cdiv(M, 64));
  gemm_wmma_kernel<<<grid, 256, 0, stream>>>(A, Wt, bias, out, M, N, K);
}

extern "C" void kernel_launch(void* const* d_in, const int* in_sizes, int n_in,
                              void* d_out, int out_size, void* d_ws, size_t ws_size,
                              hipStream_t stream) {
  (void)in_sizes; (void)n_in; (void)out_size; (void)ws_size;
  const float* x         = (const float*)d_in[0];
  const float* w_qkv_s   = (const float*)d_in[1];
  const float* b_qkv_s   = (const float*)d_in[2];
  const float* w_proj_s  = (const float*)d_in[3];
  const float* b_proj_s  = (const float*)d_in[4];
  const float* temp_p    = (const float*)d_in[5];
  const float* w_qkv_w   = (const float*)d_in[6];
  const float* b_qkv_w   = (const float*)d_in[7];
  const float* w_proj_w  = (const float*)d_in[8];
  const float* b_proj_w  = (const float*)d_in[9];
  const float* conv_w[3] = {(const float*)d_in[10], (const float*)d_in[11], (const float*)d_in[12]};
  const float* spec_w[3] = {(const float*)d_in[13], (const float*)d_in[14], (const float*)d_in[15]};
  const float* fusion_w  = (const float*)d_in[16];
  const float* fusion_b  = (const float*)d_in[17];
  const float* gn_gamma  = (const float*)d_in[18];
  const float* gn_beta   = (const float*)d_in[19];
  float* out = (float*)d_out;
  float* ws  = (float*)d_ws;

  // workspace layout (float counts)
  const size_t MW   = (size_t)NWIN_ * WT_;            // 7744 rows (window stage max)
  float* ws_t    = ws;                 // max(4608,7744)*256  (later reused as ybuf)
  float* ws_qkv  = ws_t    + MW * C_;          // max rows * 768
  float* ws_attn = ws_qkv  + MW * 3 * C_;      // max rows * 256
  float* ws_proj = ws_attn + MW * C_;          // max rows * 256
  float* ws_x1   = ws_proj + MW * C_;          // B*C*H*W
  float* ws_x2   = ws_x1   + (size_t)B_ * C_ * N_;
  float* ws_xp   = ws_x2   + (size_t)B_ * C_ * N_;     // B*256*256
  float* ws_qkvp = ws_xp   + (size_t)B_ * 256 * C_;    // B*256*279
  float* ws_osp  = ws_qkvp + (size_t)B_ * 256 * FC_;   // B*93*256
  float* ws_fus  = ws_osp  + (size_t)B_ * RD_ * 256;   // 4608*279
  float* ws_gout = ws_fus  + (size_t)B_ * N_ * FC_;    // 4608*256
  float* ws_stat = ws_gout + (size_t)B_ * N_ * C_;     // 128
  float* ws_y    = ws_t;                               // reuse after window stage

  const int elemsBCN = B_ * C_ * N_;

  // ---- Stage 1: sparse attention ----
  transpose_nc_kernel<<<cdiv(B_ * N_ * C_, 256), 256, 0, stream>>>(x, ws_t);
  launch_gemm(ws_t, w_qkv_s, b_qkv_s, ws_qkv, B_ * N_, 3 * C_, C_, stream);
  sparse_attn_kernel<<<B_ * NH_ * N_, 256, 0, stream>>>(ws_qkv, temp_p, ws_attn);
  launch_gemm(ws_attn, w_proj_s, b_proj_s, ws_proj, B_ * N_, C_, C_, stream);
  add_residual_t_kernel<<<cdiv(elemsBCN, 256), 256, 0, stream>>>(x, ws_proj, ws_x1);

  // ---- Stage 2: window attention (per batch element) ----
  for (int b = 0; b < B_; ++b) {
    gather_windows_kernel<<<cdiv(NWIN_ * WT_ * C_, 256), 256, 0, stream>>>(ws_x1, ws_t, b);
    launch_gemm(ws_t, w_qkv_w, b_qkv_w, ws_qkv, NWIN_ * WT_, 3 * C_, C_, stream);
    window_attn_kernel<<<NWIN_ * NH_, 256, 0, stream>>>(ws_qkv, ws_attn);
    launch_gemm(ws_attn, w_proj_w, b_proj_w, ws_proj, NWIN_ * WT_, C_, C_, stream);
    window_scatter_kernel<<<cdiv(C_ * H_ * W_, 256), 256, 0, stream>>>(ws_x1, ws_proj, ws_x2, b);
  }

  // ---- Stage 3: spectral ----
  const int pools[3] = {4, 8, 16};
  for (int p = 0; p < 3; ++p) {
    int s = pools[p], ss = s * s;
    pool_kernel<<<cdiv(B_ * ss * C_, 256), 256, 0, stream>>>(ws_x2, ws_xp, s);
    launch_gemm(ws_xp, conv_w[p], nullptr, ws_qkvp, B_ * ss, FC_, C_, stream);
    spec_attn_kernel<<<B_ * ss, 128, 0, stream>>>(ws_qkvp, spec_w[p], ws_osp, ss);
    resize_kernel<<<cdiv(B_ * RD_ * H_ * W_, 256), 256, 0, stream>>>(ws_osp, ws_fus, s, p * RD_);
  }
  launch_gemm(ws_fus, fusion_w, fusion_b, ws_gout, B_ * N_, C_, FC_, stream);
  fuse_residual_kernel<<<cdiv(elemsBCN, 256), 256, 0, stream>>>(ws_gout, ws_x2, ws_y);

  // ---- GroupNorm ----
  gn_stats_kernel<<<B_ * 32, 256, 0, stream>>>(ws_y, ws_stat);
  gn_apply_kernel<<<cdiv(elemsBCN, 256), 256, 0, stream>>>(ws_y, ws_stat, gn_gamma, gn_beta, out);
}
